// _S4Layer_42812234006944
// MI455X (gfx1250) — matrix operand order repaired and checked by reading, not compiled, not run on hardware
//
#include <hip/hip_runtime.h>

// ---------------------------------------------------------------------------
// S4 layer on MI455X (gfx1250): causal conv y = k * x + D.x via bf16 WMMA.
//   Pass 1 : k[d,l] = sum_s (C*dt*B)[d,s]*exp(dt_d*A_s*l), stored REVERSED +
//            zero-padded per row (krevG[d][j] = k[d][2047-j], pad j>=2048).
//   Pass 1b: transpose x (b,l,d) f32 -> xT (d,b,l) bf16.
//   Pass 2 : workgroup = 4 channels x 4 l-tiles (16 waves). Per wave:
//            acc += ToeplitzA(16x32) x Xtile(32x16) with
//            v_wmma_f32_16x16x32_bf16 (N = batch, 8 of 16 cols used),
//            manually unrolled 2x. Kernel rows staged to LDS via async
//            load-to-LDS (ASYNCcnt). Output assembled in LDS, float4 stores.
// ---------------------------------------------------------------------------

typedef __attribute__((ext_vector_type(16))) __bf16 v16bf;
typedef __attribute__((ext_vector_type(8)))  float  v8f;
typedef __attribute__((ext_vector_type(4)))  int    v4i;

typedef __attribute__((address_space(1))) v4i* g_v4i_p;   // global b128 payload
typedef __attribute__((address_space(3))) v4i* l_v4i_p;   // LDS b128 payload

#define DMODEL 512
#define SDIM   64
#define BATCH  8
#define SEQ    2048
#define KROW   2096   // reversed row + 48 zero pad, 4192 B = 262 x 16 B
#define KLDS   2112   // LDS row size (extra 16 zeroed entries)

#if defined(__has_builtin)
#if __has_builtin(__builtin_amdgcn_global_load_async_to_lds_b128)
#define HAVE_ASYNC_LDS 1
#endif
#endif

static __device__ inline unsigned short f2bf_bits(float f) {
  union { float f; unsigned u; } v; v.f = f;
  unsigned r = v.u + 0x7FFFu + ((v.u >> 16) & 1u);  // round-to-nearest-even
  return (unsigned short)(r >> 16);
}

static __device__ inline v16bf load_Atile(const unsigned* p) {
  union { v16bf v; unsigned u[8]; } A;
#pragma unroll
  for (int j = 0; j < 4; ++j) { A.u[j] = p[j]; A.u[4 + j] = p[4 + j]; }
  return A.v;
}

// ---------------- Pass 1: reversed kernel precompute (block per channel) ----
__global__ __launch_bounds__(256) void s4_kernel_precompute(
    const float* __restrict__ logA, const float* __restrict__ Bm,
    const float* __restrict__ Cm,   const float* __restrict__ logdt,
    unsigned short* __restrict__ krevG) {
  const int d   = blockIdx.x;
  const int tid = threadIdx.x;
  __shared__ float wS[SDIM], dtAS[SDIM], rS[SDIM];
  if (tid < SDIM) {
    float dt  = expf(logdt[d]);
    float A   = -expf(logA[tid]);          // HiPPO diagonal
    float dtA = dt * A;
    dtAS[tid] = dtA;
    rS[tid]   = expf(dtA);
    wS[tid]   = Cm[d * SDIM + tid] * Bm[d * SDIM + tid] * dt;  // C * B_bar
  }
  __syncthreads();
  const int l0 = tid * 8;
  float acc[8];
#pragma unroll
  for (int i = 0; i < 8; ++i) acc[i] = 0.f;
  for (int s = 0; s < SDIM; ++s) {
    float p = expf(dtAS[s] * (float)l0);   // r^l0, then recur
    float r = rS[s], w = wS[s];
#pragma unroll
    for (int i = 0; i < 8; ++i) { acc[i] += w * p; p *= r; }
  }
  unsigned short* row = krevG + (size_t)d * KROW;
#pragma unroll
  for (int i = 0; i < 8; ++i) row[2047 - (l0 + i)] = f2bf_bits(acc[i]);
  if (tid < KROW - SEQ) row[SEQ + tid] = 0;           // zero pad
}

// ---------------- Pass 1b: x (b,l,d) f32 -> xT (d,b,l) bf16 -----------------
__global__ __launch_bounds__(256) void s4_transpose_x(
    const float* __restrict__ x, unsigned short* __restrict__ xT) {
  size_t i = (size_t)blockIdx.x * 256 + threadIdx.x;   // coalesced read
  int d = (int)(i & (DMODEL - 1));
  int l = (int)((i >> 9) & (SEQ - 1));
  int b = (int)(i >> 20);
  xT[((size_t)d * BATCH + b) * SEQ + l] = f2bf_bits(x[i]);
}

// ---------------- Pass 2: Toeplitz-blocked WMMA convolution -----------------
__global__ __launch_bounds__(512) void s4_conv_wmma(
    const unsigned short* __restrict__ krevG,
    const unsigned short* __restrict__ xT,
    const float* __restrict__ x,
    const float* __restrict__ Dvec,
    float* __restrict__ out) {
  __shared__ __align__(16) unsigned short krev [4][KLDS];
  __shared__ __align__(16) unsigned short krevS[4][KLDS];  // shifted copy
  __shared__ __align__(16) float stage[4][16][8][4];       // [ltile][l][b][d]

  const int tid = threadIdx.x;
  const int d0  = blockIdx.y * 4;
  const int L0  = blockIdx.x * 64;

  // ---- stage 4 reversed kernel rows into LDS ----
  {
    const int row = tid >> 7;            // 0..3 (4 waves per row)
    const int i0  = tid & 127;
    const unsigned short* src = krevG + (size_t)(d0 + row) * KROW;
#ifdef HAVE_ASYNC_LDS
    for (int i = i0; i < KROW / 8; i += 128)
      __builtin_amdgcn_global_load_async_to_lds_b128(
          (g_v4i_p)(src + i * 8), (l_v4i_p)&krev[row][i * 8], 0, 0);
#if __has_builtin(__builtin_amdgcn_s_wait_asynccnt)
    __builtin_amdgcn_s_wait_asynccnt(0);
#else
    asm volatile("s_wait_asynccnt 0x0" ::: "memory");
#endif
#else
    for (int i = i0; i < KROW / 8; i += 128)
      ((uint4*)&krev[row][0])[i] = ((const uint4*)src)[i];
#endif
    if (i0 < (KLDS - KROW)) krev[row][KROW + i0] = 0;    // LDS pad
  }
  __syncthreads();
  {  // shifted copy so every Toeplitz base is dword-aligned in one of them
    const int row = tid >> 7;
    for (int i = tid & 127; i < KLDS - 1; i += 128) krevS[row][i] = krev[row][i + 1];
    if ((tid & 127) == 0) krevS[row][KLDS - 1] = 0;
  }
  __syncthreads();

  const int wave  = tid >> 5;
  const int lane  = tid & 31;
  const int dIdx  = wave & 3;            // channel within group
  const int lIdx  = wave >> 2;           // l-tile within group
  const int d     = d0 + dIdx;
  const int l0    = L0 + lIdx * 16;
  const int M     = lane & 15;           // A row (ISA layout)
  const int hi    = lane >> 4;
  const int koffA = hi * 16;             // A: K = koffA + e
  const int koffB = hi * 16;             // B: K = koffB + e
  const int bcol  = lane & 7;

  const unsigned short* xrow = xT + ((size_t)d * BATCH + bcol) * SEQ;
  __builtin_prefetch(xrow, 0, 3);        // global_prefetch_b8

  v8f acc;
#pragma unroll
  for (int i = 0; i < 8; ++i) acc[i] = 0.f;

  // A[M,K] = k[(l0+M)-(m0+K)]; reversed index = base + K, base steps by +32,
  // so its parity (krev vs shifted krevS row) is loop-invariant: hoist it.
  const int nsteps = (l0 + 47) / 32;     // K-chunks covering m <= l0+15
  const int base0  = 2047 - l0 - M + koffA;
  const int sel    = base0 & 1;
  const unsigned* p = (const unsigned*)(sel ? (const void*)&krevS[dIdx][base0 - 1]
                                            : (const void*)&krev [dIdx][base0]);
  const unsigned short* xp = xrow + koffB;

  // Manually unrolled 2x: both iterations' operand loads issue together and
  // overlap the accumulate-chained WMMA pair (C->D chaining, no A/B hazard).
  int s = 0;
  for (; s + 2 <= nsteps; s += 2) {
    v16bf a0 = load_Atile(p);
    v16bf a1 = load_Atile(p + 16);
    v16bf b0 = *(const v16bf*)xp;        // 32B contiguous per lane
    v16bf b1 = *(const v16bf*)(xp + 32);
    acc = __builtin_amdgcn_wmma_f32_16x16x32_bf16(
        false, a0, false, b0, (short)0, acc, false, false);
    acc = __builtin_amdgcn_wmma_f32_16x16x32_bf16(
        false, a1, false, b1, (short)0, acc, false, false);
    p  += 32;                            // +64 bf16 elements along reversed row
    xp += 64;                            // next two K-chunks of x
  }
  if (s < nsteps) {
    v16bf a0 = load_Atile(p);
    v16bf b0 = *(const v16bf*)xp;
    acc = __builtin_amdgcn_wmma_f32_16x16x32_bf16(
        false, a0, false, b0, (short)0, acc, false, false);
  }

  // C/D layout: VGPR v <-> M = v + 8*hi, N = lane&15; keep N<8 (real batches)
  if ((lane & 15) < 8) {
#pragma unroll
    for (int v = 0; v < 8; ++v) stage[lIdx][hi * 8 + v][lane & 7][dIdx] = acc[v];
  }
  __syncthreads();

  // cooperative coalesced store: float4 along d, + skip connection in f32
  {
    const int lt = tid >> 7;
    const int ll = (tid >> 3) & 15;
    const int b  = tid & 7;
    const int l  = L0 + lt * 16 + ll;
    const size_t idx = ((size_t)b * SEQ + l) * DMODEL + d0;
    float4 sv = *(const float4*)&stage[lt][ll][b][0];
    float4 xv = *(const float4*)(x + idx);
    float4 dd = *(const float4*)(Dvec + d0);
    float4 r;
    r.x = sv.x + dd.x * xv.x;
    r.y = sv.y + dd.y * xv.y;
    r.z = sv.z + dd.z * xv.z;
    r.w = sv.w + dd.w * xv.w;
    *(float4*)(out + idx) = r;
  }
}

// ---------------------------------------------------------------------------
extern "C" void kernel_launch(void* const* d_in, const int* in_sizes, int n_in,
                              void* d_out, int out_size, void* d_ws, size_t ws_size,
                              hipStream_t stream) {
  const float* x     = (const float*)d_in[0];  // (8, 2048, 512)
  const float* logA  = (const float*)d_in[1];  // (64,)
  const float* Bm    = (const float*)d_in[2];  // (512, 64)
  const float* Cm    = (const float*)d_in[3];  // (512, 64)
  const float* logdt = (const float*)d_in[4];  // (512,)
  const float* Dv    = (const float*)d_in[5];  // (512,)

  unsigned short* krevG = (unsigned short*)d_ws;                       // ~2.1 MB
  unsigned short* xT    = (unsigned short*)((char*)d_ws + (4u << 20)); // 16 MB

  s4_kernel_precompute<<<dim3(DMODEL), dim3(256), 0, stream>>>(logA, Bm, Cm, logdt, krevG);
  s4_transpose_x<<<dim3((BATCH * SEQ * DMODEL) / 256), dim3(256), 0, stream>>>(x, xT);
  s4_conv_wmma<<<dim3(SEQ / 64, DMODEL / 4), dim3(512), 0, stream>>>(
      krevG, xT, x, Dv, (float*)d_out);
}
